// RNN_arch_2_final_65670049955961
// MI455X (gfx1250) — matrix operand order repaired and checked by reading, not compiled, not run on hardware
//
#include <hip/hip_runtime.h>

typedef __bf16 bf16_t;
typedef __attribute__((ext_vector_type(16))) __bf16 v16bf;
typedef __attribute__((ext_vector_type(8)))  __bf16 v8bf;
typedef __attribute__((ext_vector_type(8)))  float  v8f;
typedef __attribute__((ext_vector_type(4)))  float  v4f;

#define B_SZ 16384
#define T_SZ 16

// ---------------- workspace layout (bf16 element offsets) ----------------
#define OW1I 0        // [256][64]
#define OW1H 16384    // [256][256]
#define OW1O 81920    // [64][256]
#define OW2I 98304
#define OW2H 114688
#define OW2O 180224
#define OW3I 196608   // [256][160] (130 padded to 160 with zeros)
#define OW3H 237568
#define OW3O 303104
#define OWFC 319488   // [16][64]  (2 rows padded to 16 with zeros)
#define WSF_BYTE_OFF 641024  // f32 bias region starts here (320512 bf16 * 2B)

// ---------------- output layout (f32 element offsets) ----------------
#define O_YS   0LL          // [16,16384,2]
#define O_H1   524288LL     // [16384,256]
#define O_H2   4718592LL
#define O_H3   8912896LL
#define O_HC4  13107200LL
#define O_O1S  17301504LL   // [16,16384,64]
#define O_O2S  34078720LL
#define O_CUE  50855936LL   // [16,16384,2]
#define O_FEED 51380224LL   // two [16,16384,256] zero tensors, contiguous
#define O_OFS  185597952LL  // [16,16384,64]

__device__ __forceinline__ v16bf comb(v8bf lo, v8bf hi) {
  return __builtin_shufflevector(lo, hi, 0,1,2,3,4,5,6,7,8,9,10,11,12,13,14,15);
}

// Branch-free tanh: tanh(x) = sign(x) * (1 - 2/(e^{2|x|}+1))
// exp2 overflow -> inf -> rcp -> 0 -> result saturates to +/-1. No EXEC games.
__device__ __forceinline__ float fast_tanh(float x) {
  float ax = fabsf(x);
  float e  = __builtin_amdgcn_exp2f(ax * 2.8853900817779268f); // 2*log2(e)
  float r  = 1.0f - 2.0f * __builtin_amdgcn_rcpf(e + 1.0f);
  return copysignf(r, x);
}

// B operand (32x16 bf16): lane L -> weight row n = nbase + (L&15),
// k at k0 + (L>>4)*8 + [0..7] and k0 + 16 + (L>>4)*8 + [0..7]
__device__ __forceinline__ v16bf ldB(const bf16_t* __restrict__ W, int ldk,
                                     int nbase, int k0, int lane) {
  const bf16_t* p = W + (nbase + (lane & 15)) * ldk + k0 + ((lane >> 4) << 3);
  return comb(*(const v8bf*)p, *(const v8bf*)(p + 16));
}

// A operand (16x32 bf16) from LDS row-major; p0 = row base (row = lane&15)
__device__ __forceinline__ v16bf ldA_lds(const bf16_t* p0, int k0, int hf) {
  const bf16_t* p = p0 + k0 + hf * 8;
  return comb(*(const v8bf*)p, *(const v8bf*)(p + 16));
}

// A operand built directly from f32 global x (row base xr), cvt to bf16
__device__ __forceinline__ v16bf ldAx(const float* __restrict__ xr, int k0, int hf) {
  const float* p = xr + k0 + hf * 8;
  v4f a0 = *(const v4f*)(p);
  v4f a1 = *(const v4f*)(p + 4);
  v4f b0 = *(const v4f*)(p + 16);
  v4f b1 = *(const v4f*)(p + 20);
  v16bf r;
  r[0]=(__bf16)a0.x;  r[1]=(__bf16)a0.y;  r[2]=(__bf16)a0.z;  r[3]=(__bf16)a0.w;
  r[4]=(__bf16)a1.x;  r[5]=(__bf16)a1.y;  r[6]=(__bf16)a1.z;  r[7]=(__bf16)a1.w;
  r[8]=(__bf16)b0.x;  r[9]=(__bf16)b0.y;  r[10]=(__bf16)b0.z; r[11]=(__bf16)b0.w;
  r[12]=(__bf16)b1.x; r[13]=(__bf16)b1.y; r[14]=(__bf16)b1.z; r[15]=(__bf16)b1.w;
  return r;
}

__device__ __forceinline__ v8f wmma_bf(v16bf a, v16bf b, v8f c) {
  return __builtin_amdgcn_wmma_f32_16x16x32_bf16(false, a, false, b, (short)0, c,
                                                 false, false);
}

// ---------------- persistent RNN kernel: 1 wave per WG, 16 rows per wave ----
__global__ __launch_bounds__(32) void rnn_main(
    const float* __restrict__ x, const float* __restrict__ cue,
    const float* __restrict__ hc1, const float* __restrict__ hc2,
    const float* __restrict__ hc3,
    const bf16_t* __restrict__ wsb, const float* __restrict__ wsf,
    float* __restrict__ out)
{
  __shared__ __align__(16) bf16_t H1[2][16*256];
  __shared__ __align__(16) bf16_t H2[2][16*256];
  __shared__ __align__(16) bf16_t H3[2][16*256];
  __shared__ __align__(16) bf16_t MID[16*160];   // [cue(2) | o1(64) | o2(64) | pad(30)]
  __shared__ __align__(16) bf16_t OFB[16*64];

  const int lane = threadIdx.x & 31;
  const int ln = lane & 15;      // A-row / C-col index
  const int hf = lane >> 4;
  const int R = blockIdx.x * 16; // batch row base for this wave

  // init persistent states (buffer 0)
  for (int idx = lane; idx < 16*256; idx += 32) {
    int r = idx >> 8, c = idx & 255;
    size_t g = (size_t)(R + r) * 256 + c;
    H1[0][idx] = (__bf16)hc1[g];
    H2[0][idx] = (__bf16)hc2[g];
    H3[0][idx] = (__bf16)hc3[g];
  }
  for (int idx = lane; idx < 16*160; idx += 32) MID[idx] = (__bf16)0.0f;
  if (lane < 16) {
    float c = cue[R + lane];
    MID[lane*160 + 0] = (__bf16)(c * 10.0f);
    MID[lane*160 + 1] = (__bf16)(10.0f * fabsf(c - 1.0f));
  }

  const bf16_t *W1I = wsb+OW1I, *W1H = wsb+OW1H, *W1O = wsb+OW1O;
  const bf16_t *W2I = wsb+OW2I, *W2H = wsb+OW2H, *W2O = wsb+OW2O;
  const bf16_t *W3I = wsb+OW3I, *W3H = wsb+OW3H, *W3O = wsb+OW3O;
  const bf16_t *WFC = wsb+OWFC;
  const float *BS1 = wsf,      *BS2 = wsf+256, *BS3 = wsf+512;
  const float *B1O = wsf+768,  *B2O = wsf+832, *B3O = wsf+896, *BFC = wsf+960;

  float* out_ys  = out + O_YS;
  float* out_h1  = out + O_H1;
  float* out_h2  = out + O_H2;
  float* out_h3  = out + O_H3;
  float* out_o1s = out + O_O1S;
  float* out_o2s = out + O_O2S;
  float* out_ofs = out + O_OFS;

  for (int t = 0; t < T_SZ; ++t) {
    const int rb = t & 1, wb = rb ^ 1;
    const bool last = (t == T_SZ - 1);
    const float* xrow = x + ((size_t)t * B_SZ + R + ln) * 64;

    // ---- layer 1 hidden: h1 = tanh(x@W1i^T + h1@W1h^T + bsum1) ----
    {
      const bf16_t* h1r = &H1[rb][ln*256];
      for (int nt = 0; nt < 16; ++nt) {
        float bias = BS1[nt*16 + ln];
        v8f acc;
        #pragma unroll
        for (int r = 0; r < 8; ++r) acc[r] = bias;
        #pragma unroll
        for (int kc = 0; kc < 2; ++kc)
          acc = wmma_bf(ldAx(xrow, kc*32, hf), ldB(W1I, 64, nt*16, kc*32, lane), acc);
        #pragma unroll
        for (int kc = 0; kc < 8; ++kc)
          acc = wmma_bf(ldA_lds(h1r, kc*32, hf), ldB(W1H, 256, nt*16, kc*32, lane), acc);
        #pragma unroll
        for (int r = 0; r < 8; ++r) {
          float v = fast_tanh(acc[r]);
          int row = r + 8*hf, col = nt*16 + ln;
          H1[wb][row*256 + col] = (__bf16)v;
          if (last) out_h1[(size_t)(R+row)*256 + col] = v;
        }
      }
    }
    // ---- o1 = tanh(h1@W1o^T + b1o) ----
    {
      const bf16_t* h1r = &H1[wb][ln*256];
      for (int nt = 0; nt < 4; ++nt) {
        float bias = B1O[nt*16 + ln];
        v8f acc;
        #pragma unroll
        for (int r = 0; r < 8; ++r) acc[r] = bias;
        #pragma unroll
        for (int kc = 0; kc < 8; ++kc)
          acc = wmma_bf(ldA_lds(h1r, kc*32, hf), ldB(W1O, 256, nt*16, kc*32, lane), acc);
        #pragma unroll
        for (int r = 0; r < 8; ++r) {
          float v = fast_tanh(acc[r]);
          int row = r + 8*hf, col = nt*16 + ln;
          out_o1s[((size_t)t*B_SZ + R + row)*64 + col] = v;
          MID[row*160 + 2 + col] = (__bf16)v;
        }
      }
    }
    // ---- layer 2 hidden ----
    {
      const bf16_t* h2r = &H2[rb][ln*256];
      for (int nt = 0; nt < 16; ++nt) {
        float bias = BS2[nt*16 + ln];
        v8f acc;
        #pragma unroll
        for (int r = 0; r < 8; ++r) acc[r] = bias;
        #pragma unroll
        for (int kc = 0; kc < 2; ++kc)
          acc = wmma_bf(ldAx(xrow, kc*32, hf), ldB(W2I, 64, nt*16, kc*32, lane), acc);
        #pragma unroll
        for (int kc = 0; kc < 8; ++kc)
          acc = wmma_bf(ldA_lds(h2r, kc*32, hf), ldB(W2H, 256, nt*16, kc*32, lane), acc);
        #pragma unroll
        for (int r = 0; r < 8; ++r) {
          float v = fast_tanh(acc[r]);
          int row = r + 8*hf, col = nt*16 + ln;
          H2[wb][row*256 + col] = (__bf16)v;
          if (last) out_h2[(size_t)(R+row)*256 + col] = v;
        }
      }
    }
    // ---- o2 ----
    {
      const bf16_t* h2r = &H2[wb][ln*256];
      for (int nt = 0; nt < 4; ++nt) {
        float bias = B2O[nt*16 + ln];
        v8f acc;
        #pragma unroll
        for (int r = 0; r < 8; ++r) acc[r] = bias;
        #pragma unroll
        for (int kc = 0; kc < 8; ++kc)
          acc = wmma_bf(ldA_lds(h2r, kc*32, hf), ldB(W2O, 256, nt*16, kc*32, lane), acc);
        #pragma unroll
        for (int r = 0; r < 8; ++r) {
          float v = fast_tanh(acc[r]);
          int row = r + 8*hf, col = nt*16 + ln;
          out_o2s[((size_t)t*B_SZ + R + row)*64 + col] = v;
          MID[row*160 + 66 + col] = (__bf16)v;
        }
      }
    }
    // ---- layer 3 hidden: h3 = tanh(mid@W3i^T + h3@W3h^T + bsum3) ----
    {
      const bf16_t* midr = &MID[ln*160];
      const bf16_t* h3r  = &H3[rb][ln*256];
      for (int nt = 0; nt < 16; ++nt) {
        float bias = BS3[nt*16 + ln];
        v8f acc;
        #pragma unroll
        for (int r = 0; r < 8; ++r) acc[r] = bias;
        #pragma unroll
        for (int kc = 0; kc < 5; ++kc)
          acc = wmma_bf(ldA_lds(midr, kc*32, hf), ldB(W3I, 160, nt*16, kc*32, lane), acc);
        #pragma unroll
        for (int kc = 0; kc < 8; ++kc)
          acc = wmma_bf(ldA_lds(h3r, kc*32, hf), ldB(W3H, 256, nt*16, kc*32, lane), acc);
        #pragma unroll
        for (int r = 0; r < 8; ++r) {
          float v = fast_tanh(acc[r]);
          int row = r + 8*hf, col = nt*16 + ln;
          H3[wb][row*256 + col] = (__bf16)v;
          if (last) out_h3[(size_t)(R+row)*256 + col] = v;
        }
      }
    }
    // ---- of = relu(h3@W3o^T + b3o) ----
    {
      const bf16_t* h3r = &H3[wb][ln*256];
      for (int nt = 0; nt < 4; ++nt) {
        float bias = B3O[nt*16 + ln];
        v8f acc;
        #pragma unroll
        for (int r = 0; r < 8; ++r) acc[r] = bias;
        #pragma unroll
        for (int kc = 0; kc < 8; ++kc)
          acc = wmma_bf(ldA_lds(h3r, kc*32, hf), ldB(W3O, 256, nt*16, kc*32, lane), acc);
        #pragma unroll
        for (int r = 0; r < 8; ++r) {
          float v = fmaxf(acc[r], 0.0f);
          int row = r + 8*hf, col = nt*16 + ln;
          out_ofs[((size_t)t*B_SZ + R + row)*64 + col] = v;
          OFB[row*64 + col] = (__bf16)v;
        }
      }
    }
    // ---- y = of@Wfc^T + bfc  (N padded 2->16) ----
    {
      const bf16_t* ofr = &OFB[ln*64];
      float bias = BFC[ln];
      v8f acc;
      #pragma unroll
      for (int r = 0; r < 8; ++r) acc[r] = bias;
      #pragma unroll
      for (int kc = 0; kc < 2; ++kc)
        acc = wmma_bf(ldA_lds(ofr, kc*32, hf), ldB(WFC, 64, 0, kc*32, lane), acc);
      if (ln < 2) {
        #pragma unroll
        for (int r = 0; r < 8; ++r)
          out_ys[((size_t)t*B_SZ + R + r + 8*hf)*2 + ln] = acc[r];
      }
    }
  }
}

// ---------------- weight prep: f32 -> bf16 (with padding), bias sums --------
__global__ void prep_kernel(
    const float* w1i, const float* b1i, const float* w1h, const float* b1h,
    const float* w1o, const float* b1o,
    const float* w2i, const float* b2i, const float* w2h, const float* b2h,
    const float* w2o, const float* b2o,
    const float* w3i, const float* b3i, const float* w3h, const float* b3h,
    const float* w3o, const float* b3o,
    const float* wfc, const float* bfc,
    bf16_t* wsb, float* wsf)
{
  int tid = blockIdx.x * blockDim.x + threadIdx.x;
  int stride = gridDim.x * blockDim.x;
  for (int i = tid; i < 16384; i += stride) wsb[OW1I+i] = (__bf16)w1i[i];
  for (int i = tid; i < 65536; i += stride) wsb[OW1H+i] = (__bf16)w1h[i];
  for (int i = tid; i < 16384; i += stride) wsb[OW1O+i] = (__bf16)w1o[i];
  for (int i = tid; i < 16384; i += stride) wsb[OW2I+i] = (__bf16)w2i[i];
  for (int i = tid; i < 65536; i += stride) wsb[OW2H+i] = (__bf16)w2h[i];
  for (int i = tid; i < 16384; i += stride) wsb[OW2O+i] = (__bf16)w2o[i];
  for (int i = tid; i < 40960; i += stride) {
    int n = i / 160, k = i % 160;
    wsb[OW3I+i] = (__bf16)((k < 130) ? w3i[n*130 + k] : 0.0f);
  }
  for (int i = tid; i < 65536; i += stride) wsb[OW3H+i] = (__bf16)w3h[i];
  for (int i = tid; i < 16384; i += stride) wsb[OW3O+i] = (__bf16)w3o[i];
  for (int i = tid; i < 1024; i += stride) {
    int n = i / 64, k = i % 64;
    wsb[OWFC+i] = (__bf16)((n < 2) ? wfc[n*64 + k] : 0.0f);
  }
  for (int i = tid; i < 256; i += stride) {
    wsf[i]       = b1i[i] + b1h[i];
    wsf[256 + i] = b2i[i] + b2h[i];
    wsf[512 + i] = b3i[i] + b3h[i];
  }
  for (int i = tid; i < 64; i += stride) {
    wsf[768 + i] = b1o[i];
    wsf[832 + i] = b2o[i];
    wsf[896 + i] = b3o[i];
  }
  for (int i = tid; i < 16; i += stride) wsf[960 + i] = (i < 2) ? bfc[i] : 0.0f;
}

// ---------------- trivial outputs: hc4 passthrough + cue_arr ----------------
__global__ void aux_kernel(const float* __restrict__ cue,
                           const float* __restrict__ hc4,
                           float* __restrict__ out)
{
  long tid = (long)blockIdx.x * blockDim.x + threadIdx.x;
  long stride = (long)gridDim.x * blockDim.x;
  const v4f* src = (const v4f*)hc4;
  v4f* dst = (v4f*)(out + O_HC4);
  for (long i = tid; i < (4194304L/4); i += stride) {
    v4f v = src[i];
    __builtin_nontemporal_store(v, dst + i);
  }
  float* ca = out + O_CUE;
  for (long i = tid; i < B_SZ; i += stride) {
    float c = cue[i];
    float v0 = c * 10.0f;
    float v1 = 10.0f * fabsf(c - 1.0f);
    #pragma unroll
    for (int t = 0; t < T_SZ; ++t) {
      ca[((long)t*B_SZ + i)*2 + 0] = v0;
      ca[((long)t*B_SZ + i)*2 + 1] = v1;
    }
  }
}

// ------- zero-fill the two 'feed' tensors (536 MB) with NT stores ----------
__global__ void fill_kernel(v4f* __restrict__ p, long n4)
{
  long tid = (long)blockIdx.x * blockDim.x + threadIdx.x;
  long stride = (long)gridDim.x * blockDim.x;
  v4f z = {0.0f, 0.0f, 0.0f, 0.0f};
  for (long i = tid; i < n4; i += stride)
    __builtin_nontemporal_store(z, p + i);
}

extern "C" void kernel_launch(void* const* d_in, const int* in_sizes, int n_in,
                              void* d_out, int out_size, void* d_ws, size_t ws_size,
                              hipStream_t stream)
{
  const float* x   = (const float*)d_in[0];
  const float* cue = (const float*)d_in[1];
  const float* hc1 = (const float*)d_in[2];
  const float* hc2 = (const float*)d_in[3];
  const float* hc3 = (const float*)d_in[4];
  const float* hc4 = (const float*)d_in[5];
  const float* w1i = (const float*)d_in[6];  const float* b1i = (const float*)d_in[7];
  const float* w1h = (const float*)d_in[8];  const float* b1h = (const float*)d_in[9];
  const float* w1o = (const float*)d_in[10]; const float* b1o = (const float*)d_in[11];
  const float* w2i = (const float*)d_in[12]; const float* b2i = (const float*)d_in[13];
  const float* w2h = (const float*)d_in[14]; const float* b2h = (const float*)d_in[15];
  const float* w2o = (const float*)d_in[16]; const float* b2o = (const float*)d_in[17];
  const float* w3i = (const float*)d_in[18]; const float* b3i = (const float*)d_in[19];
  const float* w3h = (const float*)d_in[20]; const float* b3h = (const float*)d_in[21];
  const float* w3o = (const float*)d_in[22]; const float* b3o = (const float*)d_in[23];
  const float* wfc = (const float*)d_in[24]; const float* bfc = (const float*)d_in[25];

  float* out = (float*)d_out;
  bf16_t* wsb = (bf16_t*)d_ws;
  float* wsf = (float*)((char*)d_ws + WSF_BYTE_OFF);

  prep_kernel<<<128, 256, 0, stream>>>(w1i,b1i,w1h,b1h,w1o,b1o,
                                       w2i,b2i,w2h,b2h,w2o,b2o,
                                       w3i,b3i,w3h,b3h,w3o,b3o,
                                       wfc,bfc, wsb, wsf);
  aux_kernel<<<512, 256, 0, stream>>>(cue, hc4, out);
  fill_kernel<<<4096, 256, 0, stream>>>((v4f*)(out + O_FEED), 134217728LL/4);
  rnn_main<<<B_SZ/16, 32, 0, stream>>>(x, cue, hc1, hc2, hc3, wsb, wsf, out);
}
